// EGAT_layer_86088324481815
// MI455X (gfx1250) — matrix operand order repaired and checked by reading, not compiled, not run on hardware
//
#include <hip/hip_runtime.h>
#include <hip/hip_bf16.h>

typedef __attribute__((ext_vector_type(16))) _Float16 v16h;
typedef __attribute__((ext_vector_type(8)))  float    v8f;

#define NEG_SLOPE 0.01f
#define LN_EPS 1e-5f

// ---------------------------------------------------------------------------
// WMMA fragment helpers (CDNA5 v_wmma_f32_16x16x32_f16, wave32)
// A 16x32 (MxK): lane<16 -> row=lane, elems 0..7 = K kb..kb+7, 8..15 = K kb+16..kb+23
//                lane>=16 -> row=lane-16, elems 0..7 = K kb+8..kb+15, 8..15 = K kb+24..kb+31
// B 32x16 (KxN): col = lane&15, same K order as A: elems i = W[kb + (lane>=16?8:0) + i + (i>=8?8:0)][col]
// C/D 16x16 f32: row m = r + 8*(lane>=16), col n = lane&15
// ---------------------------------------------------------------------------
__device__ __forceinline__ v16h load_a_f32(const float* __restrict__ base, int lda,
                                           int kbase, int lane) {
  const int half = lane >> 4;
  const int m    = lane & 15;
  const float* p = base + (size_t)m * lda + kbase + half * 8;  // 32B aligned
  const float4 x0 = *(const float4*)(p);
  const float4 x1 = *(const float4*)(p + 4);
  const float4 y0 = *(const float4*)(p + 16);
  const float4 y1 = *(const float4*)(p + 20);
  v16h a;
  a[0]  = (_Float16)x0.x; a[1]  = (_Float16)x0.y; a[2]  = (_Float16)x0.z; a[3]  = (_Float16)x0.w;
  a[4]  = (_Float16)x1.x; a[5]  = (_Float16)x1.y; a[6]  = (_Float16)x1.z; a[7]  = (_Float16)x1.w;
  a[8]  = (_Float16)y0.x; a[9]  = (_Float16)y0.y; a[10] = (_Float16)y0.z; a[11] = (_Float16)y0.w;
  a[12] = (_Float16)y1.x; a[13] = (_Float16)y1.y; a[14] = (_Float16)y1.z; a[15] = (_Float16)y1.w;
  return a;
}

// Cooperative fill of LDS with weight [64, ldn] pre-swizzled into B-fragment
// order: lane-fragment f = ((tk*NT + tn)*32 + lane) occupies 16 contiguous halves.
__device__ __forceinline__ void swizzle_weight(_Float16* __restrict__ sw,
                                               const float* __restrict__ Wg,
                                               int ldn, int NT, int tid, int nthreads) {
  const int nfrag = 2 * NT * 32;  // k-tiles * n-tiles * lanes
  for (int f = tid; f < nfrag; f += nthreads) {
    const int l  = f & 31;
    const int fi = f >> 5;
    const int tk = fi / NT, tn = fi % NT;
    const int col = tn * 16 + (l & 15);
    const int k0  = tk * 32 + (l >> 4) * 8;
    _Float16* d = sw + (size_t)f * 16;
#pragma unroll
    for (int i = 0; i < 16; ++i) d[i] = (_Float16)Wg[(k0 + i + (i >= 8 ? 8 : 0)) * ldn + col];
  }
}

// One contiguous 32-byte fragment load (2x ds_load_b128) per lane.
__device__ __forceinline__ v16h load_b_sw(const _Float16* __restrict__ sw, int NT,
                                          int tk, int tn, int lane) {
  return *(const v16h*)(sw + ((size_t)(tk * NT + tn) * 32 + lane) * 16);
}

__device__ __forceinline__ v8f wmma_f16(v16h a, v16h b, v8f c) {
  return __builtin_amdgcn_wmma_f32_16x16x32_f16(false, a, false, b, (short)0, c,
                                                false, false);
}

__device__ __forceinline__ float leaky(float x) {
  return x >= 0.f ? x : NEG_SLOPE * x;
}

// ---------------------------------------------------------------------------
// 0) init: smax = -inf, ssum = 0, node_agg = 0
// ---------------------------------------------------------------------------
__global__ void init_kernel(float* __restrict__ smax, float* __restrict__ ssum,
                            float* __restrict__ nagg, int N) {
  const int i = blockIdx.x * blockDim.x + threadIdx.x;
  if (i < N * 64) nagg[i] = 0.f;
  if (i < N * 4) {
    ((int*)smax)[i] = 0xFF800000;  // -inf bits
    ssum[i] = 0.f;
  }
}

// ---------------------------------------------------------------------------
// 1) fni / fnj / h = node_feat @ {W_ni, W_nj, W_node}   [N,64]x[64,64]
// ---------------------------------------------------------------------------
__global__ void node_gemm3_kernel(const float* __restrict__ X,
                                  const float* __restrict__ Wni,
                                  const float* __restrict__ Wnj,
                                  const float* __restrict__ Wnode,
                                  float* __restrict__ fni, float* __restrict__ fnj,
                                  float* __restrict__ hfe, int ntiles) {
  __shared__ __attribute__((aligned(32))) _Float16 w[3][64 * 64];
  const int tid = threadIdx.x;
  swizzle_weight(&w[0][0], Wni, 64, 4, tid, 256);
  swizzle_weight(&w[1][0], Wnj, 64, 4, tid, 256);
  swizzle_weight(&w[2][0], Wnode, 64, 4, tid, 256);
  __syncthreads();
  const int wave = tid >> 5, lane = tid & 31;
  const int tile = blockIdx.x * 8 + wave;
  if (tile >= ntiles) return;
  const int row0 = tile * 16;
  const float* base = X + (size_t)row0 * 64;
  const v16h a0 = load_a_f32(base, 64, 0, lane);
  const v16h a1 = load_a_f32(base, 64, 32, lane);
  float* outs[3] = {fni, fnj, hfe};
  const int mh = (lane >> 4) * 8;
#pragma unroll
  for (int ws = 0; ws < 3; ++ws) {
    float* O = outs[ws];
#pragma unroll
    for (int t = 0; t < 4; ++t) {
      v8f c = {};
      c = wmma_f16(a0, load_b_sw(&w[ws][0], 4, 0, t, lane), c);
      c = wmma_f16(a1, load_b_sw(&w[ws][0], 4, 1, t, lane), c);
      const int col = t * 16 + (lane & 15);
#pragma unroll
      for (int r = 0; r < 8; ++r)
        O[(size_t)(row0 + mh + r) * 64 + col] = c[r];
    }
  }
}

// ---------------------------------------------------------------------------
// 2) f_out = leaky(fni[src] + fnj[dst] + edge_feat@W_fij + bias); scores
// ---------------------------------------------------------------------------
__global__ void edge_fuse_kernel(const float* __restrict__ Ef,
                                 const float* __restrict__ Wfij,
                                 const float* __restrict__ bias,
                                 const float* __restrict__ attn,
                                 const float* __restrict__ fni,
                                 const float* __restrict__ fnj,
                                 const int* __restrict__ src,
                                 const int* __restrict__ dst,
                                 float* __restrict__ fout,
                                 float* __restrict__ scores, int ntiles) {
  __shared__ __attribute__((aligned(32))) _Float16 w[64 * 64];
  __shared__ float sb[64], sa[64];
  const int tid = threadIdx.x;
  swizzle_weight(w, Wfij, 64, 4, tid, 256);
  if (tid < 64) { sb[tid] = bias[tid]; sa[tid] = attn[tid]; }
  __syncthreads();
  const int wave = tid >> 5, lane = tid & 31;
  const int tile = blockIdx.x * 8 + wave;
  if (tile >= ntiles) return;
  const int e0 = tile * 16;
  const float* base = Ef + (size_t)e0 * 64;
  __builtin_prefetch(base + 16 * 64, 0, 1);  // next tile's edge rows
  const v16h a0 = load_a_f32(base, 64, 0, lane);
  const v16h a1 = load_a_f32(base, 64, 32, lane);
  const int half = lane >> 4, cl = lane & 15;
  int se[8], de[8];
#pragma unroll
  for (int r = 0; r < 8; ++r) {
    se[r] = src[e0 + half * 8 + r];
    de[r] = dst[e0 + half * 8 + r];
  }
#pragma unroll
  for (int t = 0; t < 4; ++t) {
    v8f c = {};
    c = wmma_f16(a0, load_b_sw(w, 4, 0, t, lane), c);
    c = wmma_f16(a1, load_b_sw(w, 4, 1, t, lane), c);
    const int col = t * 16 + cl;
    const float bcol = sb[col];
    const float acol = sa[col];
#pragma unroll
    for (int r = 0; r < 8; ++r) {
      float v = c[r] + fni[(size_t)se[r] * 64 + col] + fnj[(size_t)de[r] * 64 + col] + bcol;
      v = leaky(v);
      fout[(size_t)(e0 + half * 8 + r) * 64 + col] = v;
      float s = v * acol;  // score contribution, head t = this col tile
      s += __shfl_xor(s, 1, 16);
      s += __shfl_xor(s, 2, 16);
      s += __shfl_xor(s, 4, 16);
      s += __shfl_xor(s, 8, 16);
      if (cl == 0) scores[(size_t)(e0 + half * 8 + r) * 4 + t] = s;
    }
  }
}

// ---------------------------------------------------------------------------
// 3) segment max over dst (float via int-ordering atomics)
// ---------------------------------------------------------------------------
__global__ void segmax_kernel(const float* __restrict__ scores,
                              const int* __restrict__ dst,
                              float* __restrict__ smax, int E) {
  const int i = blockIdx.x * blockDim.x + threadIdx.x;
  if (i >= E * 4) return;
  const float s = scores[i];
  const int d = dst[i >> 2];
  int* addr = (int*)&smax[d * 4 + (i & 3)];
  if (s >= 0.f) atomicMax(addr, __float_as_int(s));
  else          atomicMin((unsigned int*)addr, __float_as_uint(s));
}

// 4) ex = exp(score - smax[dst]) in place; ssum += ex
__global__ void expsum_kernel(float* __restrict__ scores, const int* __restrict__ dst,
                              const float* __restrict__ smax, float* __restrict__ ssum,
                              int E) {
  const int i = blockIdx.x * blockDim.x + threadIdx.x;
  if (i >= E * 4) return;
  const int d = dst[i >> 2], h = i & 3;
  const float ex = __expf(scores[i] - smax[d * 4 + h]);
  scores[i] = ex;
  atomicAdd(&ssum[d * 4 + h], ex);
}

// 5) alpha = ex / ssum[dst] in place
__global__ void alpha_kernel(float* __restrict__ scores, const int* __restrict__ dst,
                             const float* __restrict__ ssum, int E) {
  const int i = blockIdx.x * blockDim.x + threadIdx.x;
  if (i >= E * 4) return;
  scores[i] = scores[i] / ssum[dst[i >> 2] * 4 + (i & 3)];
}

// 6) node_agg[dst] += h[src] * alpha   (one thread per (edge, col))
__global__ void scatter_kernel(const float* __restrict__ hfe,
                               const float* __restrict__ alpha,
                               const int* __restrict__ src,
                               const int* __restrict__ dst,
                               float* __restrict__ nagg, int E) {
  const int i = blockIdx.x * blockDim.x + threadIdx.x;
  if (i >= E * 64) return;
  const int e = i >> 6, col = i & 63;
  const float a = alpha[e * 4 + (col >> 4)];
  atomicAdd(&nagg[(size_t)dst[e] * 64 + col], hfe[(size_t)src[e] * 64 + col] * a);
}

// ---------------------------------------------------------------------------
// 7/8) out = leaky(layernorm(relu(X @ W + b)))   [R,64]x[64,16]
// ---------------------------------------------------------------------------
__global__ void out_ln_kernel(const float* __restrict__ X, const float* __restrict__ Wg,
                              const float* __restrict__ bias,
                              const float* __restrict__ gamma,
                              const float* __restrict__ beta,
                              float* __restrict__ out, int ntiles) {
  __shared__ __attribute__((aligned(32))) _Float16 w[64 * 16];
  __shared__ float sb[16], sg[16], sbt[16];
  const int tid = threadIdx.x;
  swizzle_weight(w, Wg, 16, 1, tid, 256);
  if (tid < 16) { sb[tid] = bias[tid]; sg[tid] = gamma[tid]; sbt[tid] = beta[tid]; }
  __syncthreads();
  const int wave = tid >> 5, lane = tid & 31;
  const int tile = blockIdx.x * 8 + wave;
  if (tile >= ntiles) return;
  const int row0 = tile * 16;
  const float* base = X + (size_t)row0 * 64;
  const v16h a0 = load_a_f32(base, 64, 0, lane);
  const v16h a1 = load_a_f32(base, 64, 32, lane);
  v8f c = {};
  c = wmma_f16(a0, load_b_sw(w, 1, 0, 0, lane), c);
  c = wmma_f16(a1, load_b_sw(w, 1, 1, 0, lane), c);
  const int half = lane >> 4, col = lane & 15;
  const float g = sg[col], bt = sbt[col], bb = sb[col];
#pragma unroll
  for (int r = 0; r < 8; ++r) {
    float x = c[r] + bb;
    x = x > 0.f ? x : 0.f;  // relu
    float s = x;            // mean over the 16 cols of this row (one 16-lane half)
    s += __shfl_xor(s, 1, 16);
    s += __shfl_xor(s, 2, 16);
    s += __shfl_xor(s, 4, 16);
    s += __shfl_xor(s, 8, 16);
    const float mu = s * (1.f / 16.f);
    const float d = x - mu;
    float q = d * d;
    q += __shfl_xor(q, 1, 16);
    q += __shfl_xor(q, 2, 16);
    q += __shfl_xor(q, 4, 16);
    q += __shfl_xor(q, 8, 16);
    const float rstd = rsqrtf(q * (1.f / 16.f) + LN_EPS);
    const float y = leaky(d * rstd * g + bt);
    out[(size_t)(row0 + half * 8 + r) * 16 + col] = y;
  }
}

// ---------------------------------------------------------------------------
extern "C" void kernel_launch(void* const* d_in, const int* in_sizes, int n_in,
                              void* d_out, int out_size, void* d_ws, size_t ws_size,
                              hipStream_t stream) {
  const float* node_feat = (const float*)d_in[0];
  const float* edge_feat = (const float*)d_in[1];
  const int*   src       = (const int*)d_in[2];
  const int*   dst       = (const int*)d_in[3];
  const float* W_ni      = (const float*)d_in[4];
  const float* W_nj      = (const float*)d_in[5];
  const float* W_fij     = (const float*)d_in[6];
  const float* bias_e    = (const float*)d_in[7];
  const float* attn      = (const float*)d_in[8];
  const float* W_node    = (const float*)d_in[9];
  const float* Wv        = (const float*)d_in[10];
  const float* bv        = (const float*)d_in[11];
  const float* We        = (const float*)d_in[12];
  const float* be        = (const float*)d_in[13];
  const float* gamma_n   = (const float*)d_in[14];
  const float* beta_n    = (const float*)d_in[15];
  const float* gamma_e   = (const float*)d_in[16];
  const float* beta_e    = (const float*)d_in[17];

  const int N = in_sizes[0] / 64;   // 50000
  const int E = in_sizes[2];        // 800000

  float* ws     = (float*)d_ws;
  float* fni    = ws;
  float* fnj    = fni + (size_t)N * 64;
  float* hfe    = fnj + (size_t)N * 64;
  float* fout   = hfe + (size_t)N * 64;
  float* scores = fout + (size_t)E * 64;   // reused for ex and alpha in place
  float* smax   = scores + (size_t)E * 4;
  float* ssum   = smax + (size_t)N * 4;
  float* nagg   = ssum + (size_t)N * 4;

  float* node_out = (float*)d_out;
  float* edge_out = node_out + (size_t)N * 16;

  const int ntilesN = (N + 15) / 16;
  const int ntilesE = (E + 15) / 16;

  init_kernel<<<(N * 64 + 255) / 256, 256, 0, stream>>>(smax, ssum, nagg, N);

  node_gemm3_kernel<<<(ntilesN + 7) / 8, 256, 0, stream>>>(
      node_feat, W_ni, W_nj, W_node, fni, fnj, hfe, ntilesN);

  edge_fuse_kernel<<<(ntilesE + 7) / 8, 256, 0, stream>>>(
      edge_feat, W_fij, bias_e, attn, fni, fnj, src, dst, fout, scores, ntilesE);

  segmax_kernel<<<(E * 4 + 255) / 256, 256, 0, stream>>>(scores, dst, smax, E);
  expsum_kernel<<<(E * 4 + 255) / 256, 256, 0, stream>>>(scores, dst, smax, ssum, E);
  alpha_kernel<<<(E * 4 + 255) / 256, 256, 0, stream>>>(scores, dst, ssum, E);

  scatter_kernel<<<(E * 64 + 255) / 256, 256, 0, stream>>>(hfe, scores, src, dst, nagg, E);

  out_ln_kernel<<<(ntilesN + 7) / 8, 256, 0, stream>>>(
      nagg, Wv, bv, gamma_n, beta_n, node_out, ntilesN);
  out_ln_kernel<<<(ntilesE + 7) / 8, 256, 0, stream>>>(
      fout, We, be, gamma_e, beta_e, edge_out, ntilesE);
}
